// SchNetGatedMoE_76012331204799
// MI455X (gfx1250) — matrix operand order repaired and checked by reading, not compiled, not run on hardware
//
#include <hip/hip_runtime.h>
#include <hip/hip_bf16.h>

// ---------------------------------------------------------------------------
// SchNet + gated MoE head for MI455X (gfx1250), wave32 + WMMA f16->f32.
//
// Workspace layout: x[N*64] f32 | agg[N*64] f32 | pooled[G*64] f32 |
//                   counts[G] f32 | wf16 (pre-swizzled f16 weight fragments)
//
// WMMA layouts used (CDNA5 ISA 7.12.2, wave32):
//  A 16x32 f16 : lane l holds row m=l&15; elem e -> K = (e<8)? 8*half+e
//                : 8+8*half+e  (half = l>>4)
//  B 32x16 f16 : lane l holds col n=l&15; elem e -> K = (e<8)? 8*half+e : 8+8*half+e
//  C/D 16x16 f32: lane l holds col n=l&15; vgpr v -> row m = v + 8*half
// ---------------------------------------------------------------------------

typedef __attribute__((ext_vector_type(16))) _Float16 v16h;
typedef __attribute__((ext_vector_type(8)))  _Float16 v8h;
typedef __attribute__((ext_vector_type(8)))  float    v8f;
typedef __attribute__((ext_vector_type(4)))  float    v4f;

#define HID 64
#define NRBF 32
#define NLAYERS 3
#define FRAGS_PER_LAYER 36   // fW1:4  fW2:8  dW:8  uW1:8  uW2:8
#define FRAG_HALVES 512      // 32 lanes * 16 f16

__device__ __forceinline__ v8f wmma16(v16h a, v16h b, v8f c) {
  // D = A(16x32,f16) x B(32x16,f16) + C(16x16,f32)
  return __builtin_amdgcn_wmma_f32_16x16x32_f16(false, a, false, b,
                                                (short)0, c, false, false);
}

__device__ __forceinline__ float fast_rcp(float v) {
  return __builtin_amdgcn_rcpf(v);   // single v_rcp_f32
}

__device__ __forceinline__ float silu_f(float v) {
  // x * sigmoid(x) with hw reciprocal: avoids IEEE div expansion
  return v * fast_rcp(1.0f + __expf(-v));
}

__device__ __forceinline__ void atom_add_f32(float* p, float v) {
  unsafeAtomicAdd(p, v);   // lowers to global_atomic_add_f32
}

// Load an A-fragment (16x32 f16) from a row-major f32 row (64 floats).
// K runs are contiguous in chunks of 8 -> two 16B loads per half.
__device__ __forceinline__ v16h a_frag_from_f32row(const float* __restrict__ row,
                                                   int kc, int half) {
  const float* p = row + kc * 32 + 8 * half;
  v4f x0 = *(const v4f*)(p);
  v4f x1 = *(const v4f*)(p + 4);
  v4f x2 = *(const v4f*)(p + 16);
  v4f x3 = *(const v4f*)(p + 20);
  v16h a;
#pragma unroll
  for (int j = 0; j < 4; ++j) {
    a[j]      = (_Float16)x0[j];
    a[4 + j]  = (_Float16)x1[j];
    a[8 + j]  = (_Float16)x2[j];
    a[12 + j] = (_Float16)x3[j];
  }
  return a;
}

// Load an A-fragment from a row-major f16 LDS tile row (64 halves).
__device__ __forceinline__ v16h a_frag_from_lds(const _Float16* rowbase,
                                                int kc, int half) {
  const _Float16* p = rowbase + kc * 32 + 8 * half;
  v8h lo = *(const v8h*)(p);
  v8h hi = *(const v8h*)(p + 16);
  return __builtin_shufflevector(lo, hi, 0, 1, 2, 3, 4, 5, 6, 7,
                                 8, 9, 10, 11, 12, 13, 14, 15);
}

// ---------------------------------------------------------------------------
// Pre-swizzle all layer weights (torch [out,in] f32) into f16 B-fragments.
// Fragment order per layer: fW1[nt 0..3] | fW2[kc][nt] | dW | uW1 | uW2
// ---------------------------------------------------------------------------
__global__ void prep_weights_kernel(const float* __restrict__ fW1,
                                    const float* __restrict__ fW2,
                                    const float* __restrict__ dW,
                                    const float* __restrict__ uW1,
                                    const float* __restrict__ uW2,
                                    _Float16* __restrict__ wf, int total) {
  int idx = blockIdx.x * blockDim.x + threadIdx.x;
  if (idx >= total) return;
  int layer = idx / (FRAGS_PER_LAYER * FRAG_HALVES);
  int r     = idx % (FRAGS_PER_LAYER * FRAG_HALVES);
  int frag  = r / FRAG_HALVES;
  int pos   = r % FRAG_HALVES;
  int lane  = pos >> 4;
  int e     = pos & 15;
  int n     = lane & 15;
  int hs    = lane >> 4;

  const float* W;
  int K, nt, kc;
  if (frag < 4)       { W = fW1 + layer * 64 * 32; K = 32; nt = frag;            kc = 0; }
  else if (frag < 12) { int f = frag - 4;  W = fW2 + layer * 4096; K = 64; kc = f >> 2; nt = f & 3; }
  else if (frag < 20) { int f = frag - 12; W = dW  + layer * 4096; K = 64; kc = f >> 2; nt = f & 3; }
  else if (frag < 28) { int f = frag - 20; W = uW1 + layer * 4096; K = 64; kc = f >> 2; nt = f & 3; }
  else                { int f = frag - 28; W = uW2 + layer * 4096; K = 64; kc = f >> 2; nt = f & 3; }

  int row = nt * 16 + n;                 // output-channel (B's N dim)
  int col = kc * 32 + ((e < 8) ? (8 * hs + e) : (16 + 8 * hs + (e - 8)));       // input-channel (B's K dim)
  wf[idx] = (_Float16)W[row * K + col];
}

// ---------------------------------------------------------------------------
__global__ void embed_kernel(float* __restrict__ x, const float* __restrict__ emb,
                             const int* __restrict__ z, long total) {
  long idx = (long)blockIdx.x * blockDim.x + threadIdx.x;
  if (idx >= total) return;
  long n = idx >> 6;
  int  h = (int)(idx & 63);
  x[idx] = emb[(size_t)z[n] * HID + h];
}

// ---------------------------------------------------------------------------
// Fused per-edge-tile kernel: RBF -> filter MLP (WMMA) -> gather+dense (WMMA)
// -> elementwise filter -> atomic scatter-add into agg. One wave = 16 edges.
// ---------------------------------------------------------------------------
__global__ __launch_bounds__(256) void edge_kernel(
    const float* __restrict__ x, float* __restrict__ agg,
    const int* __restrict__ eidx, const float* __restrict__ ew,
    const _Float16* __restrict__ wf,
    const float* __restrict__ fb1, const float* __restrict__ fb2,
    const float* __restrict__ db, int layer, int E, int ntiles) {
  constexpr float RBF_STEP  = 6.0f / 31.0f;
  constexpr float RBF_COEFF = -0.5f / (RBF_STEP * RBF_STEP);

  __shared__ __align__(16) _Float16 lds[8][16 * HID];
  const int wave = threadIdx.x >> 5;
  const int lane = threadIdx.x & 31;
  const int tile = blockIdx.x * 8 + wave;
  if (tile >= ntiles) return;           // no block barriers -> safe wave exit
  const int m = lane & 15, half = lane >> 4, ncol = lane & 15;
  const int eb = tile * 16;

  const _Float16* wl  = wf + (size_t)layer * FRAGS_PER_LAYER * FRAG_HALVES;
  const float* fb1l = fb1 + layer * HID;
  const float* fb2l = fb2 + layer * HID;
  const float* dbl  = db  + layer * HID;

  // ---- RBF A-fragment, computed in registers (edge row m = lane&15)
  int em = eb + m; if (em >= E) em = E - 1;
  const float wdist = ew[em];
  v16h aR;
#pragma unroll
  for (int e = 0; e < 16; ++e) {
    int K = (e < 8) ? (8 * half + e) : (8 + 8 * half + e);
    float d = wdist - (float)K * RBF_STEP;
    aR[e] = (_Float16)__expf(RBF_COEFF * d * d);
  }

  // ---- GEMM1: h1 = silu(attr @ fW1^T + fb1), write C-layout tile to LDS
  _Float16* lt = lds[wave];
#pragma unroll
  for (int nt = 0; nt < 4; ++nt) {
    v16h b = *(const v16h*)(wl + (size_t)nt * FRAG_HALVES + lane * 16);
    v8f c = {};
    c = wmma16(aR, b, c);
    float bias = fb1l[nt * 16 + ncol];
#pragma unroll
    for (int v = 0; v < 8; ++v) {
      float h = silu_f(c[v] + bias);
      lt[(v + 8 * half) * HID + nt * 16 + ncol] = (_Float16)h;
    }
  }
  asm volatile("s_wait_dscnt 0x0" ::: "memory");  // per-wave LDS RAW fence

  // ---- transpose-read h1 as A-fragments
  v16h aH[2];
#pragma unroll
  for (int kc = 0; kc < 2; ++kc) aH[kc] = a_frag_from_lds(lt + m * HID, kc, half);

  // ---- gather x[src] as A-fragments (direct, K contiguous in 8-float runs)
  const int srow = eidx[em];
  const float* xr = x + (size_t)srow * HID;
  v16h aX[2];
#pragma unroll
  for (int kc = 0; kc < 2; ++kc) aX[kc] = a_frag_from_f32row(xr, kc, half);

  // ---- destination rows for scatter (C-layout: edge row varies with v)
  int drow[8];
#pragma unroll
  for (int v = 0; v < 8; ++v) {
    int er = eb + v + 8 * half;
    drow[v] = eidx[(size_t)E + (er < E ? er : E - 1)];
  }

  // ---- GEMM2 (filter f) + GEMM3 (dense msg), multiply, scatter-add
#pragma unroll
  for (int nt = 0; nt < 4; ++nt) {
    v8f cf = {}, cm = {};
#pragma unroll
    for (int kc = 0; kc < 2; ++kc) {
      v16h bf = *(const v16h*)(wl + (size_t)(4 + kc * 4 + nt) * FRAG_HALVES + lane * 16);
      cf = wmma16(aH[kc], bf, cf);
      v16h bd = *(const v16h*)(wl + (size_t)(12 + kc * 4 + nt) * FRAG_HALVES + lane * 16);
      cm = wmma16(aX[kc], bd, cm);
    }
    const float bF = fb2l[nt * 16 + ncol];
    const float bD = dbl[nt * 16 + ncol];
#pragma unroll
    for (int v = 0; v < 8; ++v) {
      int er = eb + v + 8 * half;
      if (er < E) {
        float fv  = cf[v] + bF;
        float msg = (cm[v] + bD) * fv;
        atom_add_f32(&agg[(size_t)drow[v] * HID + nt * 16 + ncol], msg);
      }
    }
  }
}

// ---------------------------------------------------------------------------
// Node residual update: x += silu(agg @ uW1^T + ub1) @ uW2^T + ub2
// One wave = 16 nodes, same WMMA structure.
// ---------------------------------------------------------------------------
__global__ __launch_bounds__(256) void node_kernel(
    float* __restrict__ x, const float* __restrict__ agg,
    const _Float16* __restrict__ wf,
    const float* __restrict__ ub1, const float* __restrict__ ub2,
    int layer, int N, int ntiles) {
  __shared__ __align__(16) _Float16 lds[8][16 * HID];
  const int wave = threadIdx.x >> 5;
  const int lane = threadIdx.x & 31;
  const int tile = blockIdx.x * 8 + wave;
  if (tile >= ntiles) return;
  const int m = lane & 15, half = lane >> 4, ncol = lane & 15;
  const int nb = tile * 16;

  const _Float16* wl = wf + (size_t)layer * FRAGS_PER_LAYER * FRAG_HALVES;
  const float* ub1l = ub1 + layer * HID;
  const float* ub2l = ub2 + layer * HID;

  int row = nb + m; if (row >= N) row = N - 1;
  const float* ar = agg + (size_t)row * HID;
  v16h aA[2];
#pragma unroll
  for (int kc = 0; kc < 2; ++kc) aA[kc] = a_frag_from_f32row(ar, kc, half);

  _Float16* lt = lds[wave];
#pragma unroll
  for (int nt = 0; nt < 4; ++nt) {
    v8f c = {};
#pragma unroll
    for (int kc = 0; kc < 2; ++kc) {
      v16h b = *(const v16h*)(wl + (size_t)(20 + kc * 4 + nt) * FRAG_HALVES + lane * 16);
      c = wmma16(aA[kc], b, c);
    }
    float bias = ub1l[nt * 16 + ncol];
#pragma unroll
    for (int v = 0; v < 8; ++v) {
      float h = silu_f(c[v] + bias);
      lt[(v + 8 * half) * HID + nt * 16 + ncol] = (_Float16)h;
    }
  }
  asm volatile("s_wait_dscnt 0x0" ::: "memory");

  v16h aU[2];
#pragma unroll
  for (int kc = 0; kc < 2; ++kc) aU[kc] = a_frag_from_lds(lt + m * HID, kc, half);

#pragma unroll
  for (int nt = 0; nt < 4; ++nt) {
    v8f c = {};
#pragma unroll
    for (int kc = 0; kc < 2; ++kc) {
      v16h b = *(const v16h*)(wl + (size_t)(28 + kc * 4 + nt) * FRAG_HALVES + lane * 16);
      c = wmma16(aU[kc], b, c);
    }
    float bias = ub2l[nt * 16 + ncol];
#pragma unroll
    for (int v = 0; v < 8; ++v) {
      int r = nb + v + 8 * half;
      if (r < N) {
        size_t o = (size_t)r * HID + nt * 16 + ncol;
        x[o] += c[v] + bias;   // unique (r,col) per lane: no atomic needed
      }
    }
  }
}

// ---------------------------------------------------------------------------
__global__ void pool_kernel(const float* __restrict__ x, const int* __restrict__ batch,
                            float* __restrict__ pooled, float* __restrict__ counts,
                            long total) {
  long idx = (long)blockIdx.x * blockDim.x + threadIdx.x;
  if (idx >= total) return;
  long n = idx >> 6;
  int  h = (int)(idx & 63);
  int  g = batch[n];
  atom_add_f32(&pooled[(size_t)g * HID + h], x[idx]);
  if (h == 0) atom_add_f32(&counts[g], 1.0f);
}

// ---------------------------------------------------------------------------
__global__ void gate_kernel(const float* __restrict__ pooled,
                            const float* __restrict__ counts,
                            const float* __restrict__ mlip,
                            const float* __restrict__ gW1, const float* __restrict__ gb1,
                            const float* __restrict__ gW2, const float* __restrict__ gb2,
                            const float* __restrict__ gW3, const float* __restrict__ gb3,
                            float* __restrict__ out, int G) {
  int g = blockIdx.x * blockDim.x + threadIdx.x;
  if (g >= G) return;
  const float inv = 1.0f / fmaxf(counts[g], 1.0f);  // once per graph: keep IEEE
  const float* pg = pooled + (size_t)g * HID;
  const float* mg = mlip + (size_t)g * 8;

  float h1[64];
  for (int o = 0; o < 64; ++o) {
    const float* wr = gW1 + o * 72;
    float s = gb1[o];
    for (int k = 0; k < 64; ++k) s += wr[k] * (pg[k] * inv);
    for (int k = 0; k < 8; ++k)  s += wr[64 + k] * mg[k];
    h1[o] = fmaxf(s, 0.0f);
  }
  float h2[32];
  for (int o = 0; o < 32; ++o) {
    const float* wr = gW2 + o * 64;
    float s = gb2[o];
    for (int k = 0; k < 64; ++k) s += wr[k] * h1[k];
    h2[o] = fmaxf(s, 0.0f);
  }
  float lg[8];
  for (int o = 0; o < 8; ++o) {
    const float* wr = gW3 + o * 32;
    float s = gb3[o];
    for (int k = 0; k < 32; ++k) s += wr[k] * h2[k];
    lg[o] = s;
  }
  float mx = lg[0];
  for (int e = 1; e < 8; ++e) mx = fmaxf(mx, lg[e]);
  float ex[8], se = 0.0f;
  for (int e = 0; e < 8; ++e) { ex[e] = __expf(lg[e] - mx); se += ex[e]; }
  const float inv_se = fast_rcp(se);
  float pred = 0.0f;
  for (int e = 0; e < 8; ++e) {
    float w = ex[e] * inv_se;
    out[(size_t)g * 8 + e] = lg[e];                 // logits
    out[(size_t)G * 8 + (size_t)g * 8 + e] = w;     // weights
    pred += mg[e] * w;
  }
  out[(size_t)G * 16 + g] = pred;                   // prediction
}

// ---------------------------------------------------------------------------
extern "C" void kernel_launch(void* const* d_in, const int* in_sizes, int n_in,
                              void* d_out, int out_size, void* d_ws, size_t ws_size,
                              hipStream_t stream) {
  const int*   z     = (const int*)  d_in[0];
  const int*   eidx  = (const int*)  d_in[1];
  const float* ew    = (const float*)d_in[2];
  const int*   batch = (const int*)  d_in[3];
  const float* mlip  = (const float*)d_in[4];
  const float* emb   = (const float*)d_in[5];
  const float* fW1   = (const float*)d_in[6];
  const float* fb1   = (const float*)d_in[7];
  const float* fW2   = (const float*)d_in[8];
  const float* fb2   = (const float*)d_in[9];
  const float* dW    = (const float*)d_in[10];
  const float* db    = (const float*)d_in[11];
  const float* uW1   = (const float*)d_in[12];
  const float* ub1   = (const float*)d_in[13];
  const float* uW2   = (const float*)d_in[14];
  const float* ub2   = (const float*)d_in[15];
  const float* gW1   = (const float*)d_in[16];
  const float* gb1   = (const float*)d_in[17];
  const float* gW2   = (const float*)d_in[18];
  const float* gb2   = (const float*)d_in[19];
  const float* gW3   = (const float*)d_in[20];
  const float* gb3   = (const float*)d_in[21];

  const int N = in_sizes[0];
  const int E = in_sizes[2];
  const int G = in_sizes[4] / 8;

  char* ws = (char*)d_ws;
  float* x      = (float*)ws; ws += (size_t)N * HID * sizeof(float);
  float* agg    = (float*)ws; ws += (size_t)N * HID * sizeof(float);
  float* pooled = (float*)ws; ws += (size_t)G * HID * sizeof(float);
  float* counts = (float*)ws; ws += (size_t)G * sizeof(float);
  _Float16* wf  = (_Float16*)ws;

  float* out = (float*)d_out;

  {
    int total = NLAYERS * FRAGS_PER_LAYER * FRAG_HALVES;
    prep_weights_kernel<<<(total + 255) / 256, 256, 0, stream>>>(fW1, fW2, dW, uW1, uW2, wf, total);
  }
  {
    long total = (long)N * HID;
    embed_kernel<<<(int)((total + 255) / 256), 256, 0, stream>>>(x, emb, z, total);
  }

  const int etiles = (E + 15) / 16;
  const int vtiles = (N + 15) / 16;
  for (int layer = 0; layer < NLAYERS; ++layer) {
    hipMemsetAsync(agg, 0, (size_t)N * HID * sizeof(float), stream);
    edge_kernel<<<(etiles + 7) / 8, 256, 0, stream>>>(x, agg, eidx, ew, wf,
                                                      fb1, fb2, db, layer, E, etiles);
    node_kernel<<<(vtiles + 7) / 8, 256, 0, stream>>>(x, agg, wf, ub1, ub2,
                                                      layer, N, vtiles);
  }

  hipMemsetAsync(pooled, 0, (size_t)G * (HID + 1) * sizeof(float), stream);
  {
    long total = (long)N * HID;
    pool_kernel<<<(int)((total + 255) / 256), 256, 0, stream>>>(x, batch, pooled, counts, total);
  }
  gate_kernel<<<(G + 255) / 256, 256, 0, stream>>>(pooled, counts, mlip,
                                                   gW1, gb1, gW2, gb2, gW3, gb3, out, G);
}
